// GraphSAGEAutoencoder_77421080477948
// MI455X (gfx1250) — matrix operand order repaired and checked by reading, not compiled, not run on hardware
//
#include <hip/hip_runtime.h>

// Problem constants (match reference)
#define NODES 50000
#define EDGES 200000
#define FDIM  394
#define INDIM 788
#define HD    192
#define EMBD  128

typedef __attribute__((ext_vector_type(16))) __bf16 v16bf;
typedef __attribute__((ext_vector_type(8)))  float  v8f;
typedef __attribute__((ext_vector_type(2)))  __bf16 bf16x2;

struct U32x8 { uint4 lo, hi; };   // 32B carrier for a v16bf fragment

__device__ __forceinline__ unsigned pk2(float lo, float hi) {
    bf16x2 v;
    v[0] = (__bf16)lo;            // native f32->bf16 cvt (RNE)
    v[1] = (__bf16)hi;
    return __builtin_bit_cast(unsigned, v);
}

// ---------------- graph aggregation kernels ----------------

__global__ void fill0_k(float* __restrict__ p, long long n) {
    long long i = (long long)blockIdx.x * blockDim.x + threadIdx.x;
    if (i < n) p[i] = 0.0f;
}

__global__ void zero_h2_k(float* __restrict__ emb) {
    long long i = (long long)blockIdx.x * blockDim.x + threadIdx.x;
    if (i < (long long)NODES * FDIM) {
        long long node = i / FDIM;
        int c = (int)(i % FDIM);
        emb[node * INDIM + FDIM + c] = 0.0f;
    }
}

__global__ void deg_k(const int* __restrict__ dst, float* __restrict__ deg) {
    int e = blockIdx.x * blockDim.x + threadIdx.x;
    if (e < EDGES) atomicAdd(&deg[dst[e]], 1.0f);
}

__global__ void inv_k(const float* __restrict__ deg, float* __restrict__ inv) {
    int i = blockIdx.x * blockDim.x + threadIdx.x;
    if (i < NODES) inv[i] = 1.0f / fmaxf(deg[i], 1.0f);
}

__global__ void scat_k(const float* __restrict__ srcfeat, int srcld,
                       const int* __restrict__ src, const int* __restrict__ dst,
                       float* __restrict__ out, int outld, int outoff) {
    long long idx = (long long)blockIdx.x * blockDim.x + threadIdx.x;
    if (idx < (long long)EDGES * FDIM) {
        long long e = idx / FDIM;
        int c = (int)(idx % FDIM);
        float v = srcfeat[(long long)src[e] * srcld + c];
        atomicAdd(&out[(long long)dst[e] * outld + outoff + c], v);
    }
}

__global__ void scale1_k(float* __restrict__ h1, const float* __restrict__ inv) {
    long long idx = (long long)blockIdx.x * blockDim.x + threadIdx.x;
    if (idx < (long long)NODES * FDIM) {
        long long node = idx / FDIM;
        int c = (int)(idx % FDIM);
        float v = h1[idx] * inv[node];
        h1[idx] = (c == 0) ? 0.0f : v;
    }
}

__global__ void finemb_k(float* __restrict__ emb, const float* __restrict__ feat,
                         const float* __restrict__ inv) {
    long long idx = (long long)blockIdx.x * blockDim.x + threadIdx.x;
    if (idx < (long long)NODES * FDIM) {
        long long node = idx / FDIM;
        int c = (int)(idx % FDIM);
        emb[node * INDIM + c] = (c == 0) ? 0.0f : feat[node * FDIM + c];
        long long j = node * INDIM + FDIM + c;
        float v = emb[j] * inv[node];
        emb[j] = (c == 0) ? 0.0f : v;
    }
}

// ---- W[K][N] f32  ->  Wt[N][Kp] bf16 (transposed, K zero-padded to Kp) ----
__global__ void wcvt_k(const float* __restrict__ W, unsigned short* __restrict__ Wt,
                       int K, int N, int Kp) {
    long long i = (long long)blockIdx.x * blockDim.x + threadIdx.x;
    if (i < (long long)N * Kp) {
        int n = (int)(i / Kp), k = (int)(i % Kp);
        float v = (k < K) ? W[(size_t)k * N + n] : 0.0f;
        __bf16 b = (__bf16)v;
        Wt[i] = __builtin_bit_cast(unsigned short, b);
    }
}

// ---------------- WMMA bf16 GEMM (templated shapes) ----------------
// C[M,N] = act(A[M,K] @ W[K,N] + bias). Block = 128 thr = 4 waves, tile 16x64.
// Wt is pre-converted n-major bf16 (K padded). A is f32 (layer 1) or bf16.
template<int K, int N, int LDA, bool AF32, bool RELU, bool OUT32, bool OUT16, int LDC>
__global__ void __launch_bounds__(128)
gemm_wmma_t(const void* __restrict__ Ain, const unsigned short* __restrict__ Wt,
            const float* __restrict__ bias, float* __restrict__ C32,
            unsigned short* __restrict__ C16)
{
    constexpr int KP  = ((K + 31) / 32) * 32;   // Wt row stride
    constexpr int AST = 40;                     // LDS A row stride (bf16, 80B)
    constexpr int BST = 40;                     // LDS B column stride (bf16)
    constexpr int NT  = KP / 32;

    __shared__ __align__(16) unsigned short Asu[2][16 * AST];
    __shared__ __align__(16) unsigned short Bsu[2][64 * BST];

    const int tid  = threadIdx.x;
    const int wid  = tid >> 5;
    const int lane = tid & 31;
    const int half = lane >> 4;
    const int r16  = lane & 15;

    const int rowBase = blockIdx.y * 16;        // M % 16 == 0
    const int n0      = blockIdx.x * 64;
    const int col     = n0 + wid * 16 + r16;

    const int a_rr = tid >> 3;                  // 0..15 row
    const int a_kq = tid & 7;                   // 0..7  k-quad

    auto stage = [&](int it, int buf) {
        const int k0 = it * 32;
        // ---- A tile: 16 rows x 32 k
        if constexpr (AF32) {
            const float* A = (const float*)Ain;
            const int k = k0 + a_kq * 4;
            float4 v = {0.f, 0.f, 0.f, 0.f};
            if ((K % 32 == 0) || (k < K))       // K%4==0 -> quad fully valid
                v = *(const float4*)&A[(size_t)(rowBase + a_rr) * LDA + k];
            uint2 pr; pr.x = pk2(v.x, v.y); pr.y = pk2(v.z, v.w);
            *(uint2*)&Asu[buf][a_rr * AST + a_kq * 4] = pr;
        } else {
            const unsigned short* A = (const unsigned short*)Ain;  // [M][LDA] bf16
            *(uint2*)&Asu[buf][a_rr * AST + a_kq * 4] =
                *(const uint2*)&A[(size_t)(rowBase + a_rr) * LDA + k0 + a_kq * 4];
        }
        // ---- B tile: 64 n-rows x 32 k of pre-transposed bf16, pure copy
#pragma unroll
        for (int s = 0; s < 2; ++s) {
            const int u = tid + s * 128;        // 0..255
            const int nn = u >> 2, part = u & 3;
            uint4 v = {0u, 0u, 0u, 0u};
            if ((N % 64 == 0) || (n0 + nn < N))
                v = *(const uint4*)&Wt[(size_t)(n0 + nn) * KP + k0 + part * 8];
            *(uint4*)&Bsu[buf][nn * BST + part * 8] = v;
        }
    };

    v8f acc = {0.f, 0.f, 0.f, 0.f, 0.f, 0.f, 0.f, 0.f};

    stage(0, 0);
    for (int it = 0; it < NT; ++it) {
        __syncthreads();
        const int buf = it & 1;

        // a-fragment: A 16x32 layout -> lane-half 0: K {0..7,16..23}, half 1: {8..15,24..31}
        const unsigned short* ap = &Asu[buf][r16 * AST];
        U32x8 ta;
        ta.lo = *(const uint4*)(ap + half * 8);
        ta.hi = *(const uint4*)(ap + 16 + half * 8);
        v16bf a = __builtin_bit_cast(v16bf, ta);

        // b-fragment: B 32x16 layout -> lane col, half 0: K {0..7,16..23}, half 1: {8..15,24..31}
        const unsigned short* bp = &Bsu[buf][(wid * 16 + r16) * BST + half * 8];
        U32x8 tb;
        tb.lo = *(const uint4*)(bp);
        tb.hi = *(const uint4*)(bp + 16);
        v16bf b = __builtin_bit_cast(v16bf, tb);

        if (it + 1 < NT) stage(it + 1, buf ^ 1);   // global loads overlap WMMA

        acc = __builtin_amdgcn_wmma_f32_16x16x32_bf16(
            false, a, false, b, (short)0, acc, false, false);
    }

    const bool colok = (N % 64 == 0) || (col < N);
    const float bv = colok ? bias[col] : 0.0f;
#pragma unroll
    for (int v = 0; v < 8; ++v) {
        const int m = rowBase + v + half * 8;      // C/D VGPR v -> row v (+8 upper lanes)
        float r = acc[v] + bv;
        if (RELU) r = fmaxf(r, 0.0f);
        if (colok) {
            if constexpr (OUT32) C32[(size_t)m * LDC + col] = r;
            if constexpr (OUT16) {
                __bf16 h = (__bf16)r;
                C16[(size_t)m * LDC + col] = __builtin_bit_cast(unsigned short, h);
            }
        }
    }
}

// ---------------- launch ----------------

static inline unsigned cdivu(long long a, long long b) { return (unsigned)((a + b - 1) / b); }

extern "C" void kernel_launch(void* const* d_in, const int* in_sizes, int n_in,
                              void* d_out, int out_size, void* d_ws, size_t ws_size,
                              hipStream_t stream)
{
    const float* features = (const float*)d_in[0];
    const int*   src      = (const int*)d_in[1];
    const int*   dst      = (const int*)d_in[2];
    const float* W_enc1   = (const float*)d_in[3];
    const float* b_enc1   = (const float*)d_in[4];
    const float* W_enc3   = (const float*)d_in[5];
    const float* b_enc3   = (const float*)d_in[6];
    const float* W_dec1   = (const float*)d_in[7];
    const float* b_dec1   = (const float*)d_in[8];
    const float* W_dec3   = (const float*)d_in[9];
    const float* b_dec3   = (const float*)d_in[10];

    float* out = (float*)d_out;
    float* enc = out;                                  // [N,128]
    float* dec = out + (size_t)NODES * EMBD;           // [N,788]
    float* emb = dec + (size_t)NODES * INDIM;          // [N,788]

    float* ws  = (float*)d_ws;
    float* deg = ws;                                   // [N]
    float* inv = ws + NODES;                           // [N]
    // pre-converted transposed bf16 weights (K padded to 32)
    unsigned short* wt1  = (unsigned short*)(ws + 2 * (size_t)NODES);
    unsigned short* wt3  = wt1  + (size_t)HD    * 800; // [192][800]
    unsigned short* wtd1 = wt3  + (size_t)EMBD  * HD;  // [128][192]
    unsigned short* wtd3 = wtd1 + (size_t)HD    * EMBD;// [192][128]
    float* h1 = (float*)(wtd3 + (size_t)INDIM * HD);   // [N,394] f32 (16B-aligned)
    // bf16 activations reuse the h1 region once h1 is dead
    unsigned short* xb   = (unsigned short*)h1;        // [N,192] bf16
    unsigned short* encb = xb + (size_t)NODES * HD;    // [N,128] bf16
    unsigned short* yb   = encb + (size_t)NODES * EMBD;// [N,192] bf16

    const int T = 256;
    const long long NF = (long long)NODES * FDIM;
    const long long EF = (long long)EDGES * FDIM;

    fill0_k  <<<cdivu(NODES, T), T, 0, stream>>>(deg, NODES);
    fill0_k  <<<cdivu(NF, T),    T, 0, stream>>>(h1, NF);
    zero_h2_k<<<cdivu(NF, T),    T, 0, stream>>>(emb);

    // weight conversion (tiny; overlaps nothing it depends on)
    wcvt_k<<<cdivu((long long)HD   * 800, T), T, 0, stream>>>(W_enc1, wt1,  INDIM, HD,   800);
    wcvt_k<<<cdivu((long long)EMBD * HD,  T), T, 0, stream>>>(W_enc3, wt3,  HD,    EMBD, HD);
    wcvt_k<<<cdivu((long long)HD   * EMBD,T), T, 0, stream>>>(W_dec1, wtd1, EMBD,  HD,   EMBD);
    wcvt_k<<<cdivu((long long)INDIM* HD,  T), T, 0, stream>>>(W_dec3, wtd3, HD,    INDIM,HD);

    deg_k<<<cdivu(EDGES, T), T, 0, stream>>>(dst, deg);
    inv_k<<<cdivu(NODES, T), T, 0, stream>>>(deg, inv);

    scat_k  <<<cdivu(EF, T), T, 0, stream>>>(features, FDIM, src, dst, h1, FDIM, 0);
    scale1_k<<<cdivu(NF, T), T, 0, stream>>>(h1, inv);
    scat_k  <<<cdivu(EF, T), T, 0, stream>>>(h1, FDIM, src, dst, emb, INDIM, FDIM);
    finemb_k<<<cdivu(NF, T), T, 0, stream>>>(emb, features, inv);

    // autoencoder: 4 WMMA GEMMs (M = 50000 is a multiple of 16)
    const unsigned gm = NODES / 16;
    //            K      N     LDA    AF32  RELU  OUT32  OUT16  LDC
    gemm_wmma_t<INDIM, HD,   INDIM, true,  true,  false, true,  HD   >
        <<<dim3(cdivu(HD, 64),    gm), 128, 0, stream>>>(emb,  wt1,  b_enc1, nullptr, xb);
    gemm_wmma_t<HD,    EMBD, HD,    false, false, true,  true,  EMBD >
        <<<dim3(cdivu(EMBD, 64),  gm), 128, 0, stream>>>(xb,   wt3,  b_enc3, enc,     encb);
    gemm_wmma_t<EMBD,  HD,   EMBD,  false, true,  false, true,  HD   >
        <<<dim3(cdivu(HD, 64),    gm), 128, 0, stream>>>(encb, wtd1, b_dec1, nullptr, yb);
    gemm_wmma_t<HD,    INDIM,HD,    false, false, true,  false, INDIM>
        <<<dim3(cdivu(INDIM, 64), gm), 128, 0, stream>>>(yb,   wtd3, b_dec3, dec,     nullptr);
}
